// N_aryPrimitivesNN_45938970198104
// MI455X (gfx1250) — matrix operand order repaired and checked by reading, not compiled, not run on hardware
//
#include <hip/hip_runtime.h>
#include <hip/hip_bf16.h>
#include <math.h>

typedef _Float16 v16h __attribute__((ext_vector_type(16)));
typedef float    v8f  __attribute__((ext_vector_type(8)));
typedef __fp16   pkh2 __attribute__((ext_vector_type(2)));

#define DEVINL __device__ __forceinline__

union V16H { v16h v; _Float16 h[16]; unsigned u[8]; };

DEVINL unsigned pk16(float a, float b) {
  union { pkh2 h; unsigned u; } c;
  c.h = __builtin_amdgcn_cvt_pkrtz(a, b);  // a -> [15:0], b -> [31:16]
  return c.u;
}

// Packed f16 ReLU on both halves in ONE VALU op; inline asm so the compiler
// cannot reintroduce canonicalize ops (V_PK_MAX_NUM_F16, CDNA5 VOP3P op 28).
DEVINL unsigned relu_pk(unsigned x) {
  unsigned r;
  asm("v_pk_max_num_f16 %0, %1, 0" : "=v"(r) : "v"(x));
  return r;
}

// Cooperatively pack one W^T A-operand (16 out-features x K=32 chunk) into a
// 16-bit A 16x32 lane layout in contiguous k order (not AMD's table order), f16, 512 halves: dst[lane*16 + h].
// lane L: row m = L&15 -> feature f0+m; kb = k0 + (L<16?0:16);
// halves 0..15 = K kb..kb+15. W row-major [K][F].
DEVINL void fill_op(_Float16* dst, const float* __restrict__ W, int F, int f0, int k0,
                    bool padF, int tid) {
  for (int e = tid; e < 512; e += 256) {
    int lane = e >> 4;
    int h = e & 15;
    int kb = k0 + ((lane & 16) ? 16 : 0);
    int k = kb + h;
    int m = lane & 15;
    float v = 0.f;
    if (!padF || m < 8) v = W[k * F + f0 + m];
    dst[e] = (_Float16)v;
  }
}

// Per-lane bias vector matching f32 C/D layout: row m = v + (L<16?0:8), value b[f0+m].
DEVINL v8f load_bias(const float* __restrict__ Bv, int f0, int lane, bool valid) {
  v8f r;
  int base = f0 + ((lane & 16) ? 8 : 0);
#pragma unroll
  for (int v = 0; v < 8; ++v) r[v] = valid ? Bv[base + v] : 0.f;
  return r;
}

// B-operand x^T (32x16) from f16 states in LDS: lane L, col = L&15 = sample;
// halves = K (L<16 ? 0..15 : 16..31): K 0..15 = states[i] (broadcast on low lanes),
// K 16..31 = states[j0 + sample] on high lanes. One contiguous 32B read per lane.
DEVINL v16h load_xh(const _Float16* S, int i, int j0, int lane) {
  int row = (lane & 16) ? (j0 + (lane & 15)) : i;
  return *(const v16h*)(S + row * 16);
}

// Pack 8 f32 accs into 4 dwords of f16 pairs, packed-ReLU them (cvt is monotonic so
// relu-after-pack == relu-before-pack), and fetch the lane^16 partner copy.
DEVINL void pack_relu(v8f a, unsigned* own, unsigned* sw) {
#pragma unroll
  for (int p = 0; p < 4; ++p) {
    own[p] = relu_pk(pk16(a[2 * p], a[2 * p + 1]));
    // ds_swizzle SWAPX16: xor_mask=0x10, and_mask=0x1f -> offset 0x401F (lane <-> lane+16)
    sw[p] = (unsigned)__builtin_amdgcn_ds_swizzle((int)own[p], 0x401F);
  }
}

// Assemble next-layer B operand (K = 32 features) from two 16-feature D tiles.
// low lanes:  halves 0..7 = own(tileL), 8..15 = partner(tileL)
// high lanes: halves 0..7 = partner(tileH), 8..15 = own(tileH)
DEVINL v16h mixB(const unsigned* ownL, const unsigned* swL,
                 const unsigned* ownH, const unsigned* swH, bool lo) {
  V16H r;
#pragma unroll
  for (int p = 0; p < 4; ++p) {
    r.u[p]     = lo ? ownL[p] : swH[p];
    r.u[p + 4] = lo ? swL[p]  : ownH[p];
  }
  return r.v;
}

DEVINL float sigf(float x) { return 1.0f / (1.0f + __expf(-x)); }

__global__ __launch_bounds__(256) void pairmlp_kernel(
    const float* __restrict__ g_states,
    const float* __restrict__ gW1, const float* __restrict__ gb1,
    const float* __restrict__ gW2, const float* __restrict__ gb2,
    const float* __restrict__ gW3, const float* __restrict__ gb3,
    float* __restrict__ out) {
  // 14 pre-packed A-operands (512 f16 each) + f16 states tile.
  __shared__ __align__(32) _Float16 sOps[14 * 512];
  __shared__ __align__(32) _Float16 sSh[32 * 16];

  const int tid = threadIdx.x;
  const int bl = blockIdx.x;  // fused (b, l) index, 512 blocks

  // Op order: 0..3 = W1^T tiles (k0=0); 4..7 = W2^T k0=0; 8..11 = W2^T k0=32;
  //           12 = W3^T k0=0 (rows 8..15 zero); 13 = W3^T k0=32.
#pragma unroll
  for (int t = 0; t < 4; ++t) {
    fill_op(sOps + (0 + t) * 512,  gW1, 64, t * 16, 0,  false, tid);
    fill_op(sOps + (4 + t) * 512,  gW2, 64, t * 16, 0,  false, tid);
    fill_op(sOps + (8 + t) * 512,  gW2, 64, t * 16, 32, false, tid);
  }
  fill_op(sOps + 12 * 512, gW3, 8, 0, 0,  true, tid);
  fill_op(sOps + 13 * 512, gW3, 8, 0, 32, true, tid);

  for (int k = tid; k < 32 * 16; k += 256)
    sSh[k] = (_Float16)g_states[(size_t)bl * 512 + k];
  __syncthreads();

  const int lane = tid & 31;
  const int wave = tid >> 5;
  const bool lo = !(lane & 16);

  // Each lane grabs its 32B slice of every operand (contiguous LDS reads).
  const v16h* opv = (const v16h*)sOps;  // 32 v16h per operand
  v16h A1[4], A2k0[4], A2k1[4];
#pragma unroll
  for (int t = 0; t < 4; ++t) {
    A1[t]   = opv[(0 + t) * 32 + lane];
    A2k0[t] = opv[(4 + t) * 32 + lane];
    A2k1[t] = opv[(8 + t) * 32 + lane];
  }
  v16h A3_0 = opv[12 * 32 + lane];
  v16h A3_1 = opv[13 * 32 + lane];

  // Biases as WMMA C-operand initializers (broadcast reads from global, L2-hot).
  v8f bias1[4], bias2[4];
#pragma unroll
  for (int t = 0; t < 4; ++t) {
    bias1[t] = load_bias(gb1, t * 16, lane, true);
    bias2[t] = load_bias(gb2, t * 16, lane, true);
  }
  v8f bias3 = load_bias(gb3, 0, lane, lo);

  // 64 row-tiles of 16 pairs per block; 8 per wave. tile t: i = t>>1, j0 = (t&1)*16.
  for (int q = 0; q < 8; ++q) {
    int t = wave * 8 + q;
    int i = t >> 1;
    int j0 = (t & 1) << 4;

    v16h bx = load_xh(sSh, i, j0, lane);

    unsigned own[4][4], sw[4][4];

    // Layer 1: h1^T = relu(W1^T @ x^T + b1), four 16-feature tiles, K=32 in one WMMA each.
#pragma unroll
    for (int tt = 0; tt < 4; ++tt) {
      v8f a = __builtin_amdgcn_wmma_f32_16x16x32_f16(false, A1[tt], false, bx,
                                                     (short)0, bias1[tt], false, false);
      pack_relu(a, own[tt], sw[tt]);
    }
    v16h B0 = mixB(own[0], sw[0], own[1], sw[1], lo);  // h1 features 0..31 as K
    v16h B1 = mixB(own[2], sw[2], own[3], sw[3], lo);  // h1 features 32..63 as K

    // Layer 2: h2^T = relu(W2^T @ h1^T + b2), K=64 via two chained WMMAs per tile.
#pragma unroll
    for (int tt = 0; tt < 4; ++tt) {
      v8f a = __builtin_amdgcn_wmma_f32_16x16x32_f16(false, A2k0[tt], false, B0,
                                                     (short)0, bias2[tt], false, false);
      a = __builtin_amdgcn_wmma_f32_16x16x32_f16(false, A2k1[tt], false, B1,
                                                 (short)0, a, false, false);
      pack_relu(a, own[tt], sw[tt]);
    }
    v16h C0 = mixB(own[0], sw[0], own[1], sw[1], lo);
    v16h C1 = mixB(own[2], sw[2], own[3], sw[3], lo);

    // Layer 3: out^T = W3^T @ h2^T + b3 (8 valid rows, padded to 16).
    v8f o = __builtin_amdgcn_wmma_f32_16x16x32_f16(false, A3_0, false, C0,
                                                   (short)0, bias3, false, false);
    o = __builtin_amdgcn_wmma_f32_16x16x32_f16(false, A3_1, false, C1,
                                               (short)0, o, false, false);

    // Low lanes hold all 8 outputs (rows 0..7) of their sample (= column): sigmoid + store.
    if (lane < 16) {
      size_t ridx = (((size_t)bl * 32 + (size_t)i) * 32 + (size_t)(j0 + lane)) * 8;
      float4 r0, r1;
      r0.x = sigf(o[0]); r0.y = sigf(o[1]); r0.z = sigf(o[2]); r0.w = sigf(o[3]);
      r1.x = sigf(o[4]); r1.y = sigf(o[5]); r1.z = sigf(o[6]); r1.w = sigf(o[7]);
      float4* dst = (float4*)(out + ridx);
      dst[0] = r0;
      dst[1] = r1;
    }
  }
}

extern "C" void kernel_launch(void* const* d_in, const int* in_sizes, int n_in,
                              void* d_out, int out_size, void* d_ws, size_t ws_size,
                              hipStream_t stream) {
  (void)n_in; (void)out_size; (void)d_ws; (void)ws_size;
  const float* states = (const float*)d_in[0];
  const float* W1 = (const float*)d_in[1];
  const float* b1 = (const float*)d_in[2];
  const float* W2 = (const float*)d_in[3];
  const float* b2 = (const float*)d_in[4];
  const float* W3 = (const float*)d_in[5];
  const float* b3 = (const float*)d_in[6];
  float* out = (float*)d_out;

  int BL = in_sizes[0] / (32 * 16);  // B*L = 512 for the reference shapes
  pairmlp_kernel<<<BL, 256, 0, stream>>>(states, W1, b1, W2, b2, W3, b3, out);
}
